// NLBlock_46660524704005
// MI455X (gfx1250) — matrix operand order repaired and checked by reading, not compiled, not run on hardware
//
#include <hip/hip_runtime.h>
#include <hip/hip_bf16.h>

// ---------------- problem constants ----------------
#define B_   8
#define C_   256
#define IC_  128
#define N_   6272   // T*H*W = 8*28*28
#define M_   784    // pooled 4*14*14
#define MP_  800    // M padded to multiple of 32 for WMMA K loops

typedef __attribute__((ext_vector_type(16))) __bf16 v16bf;
typedef __attribute__((ext_vector_type(8)))  __bf16 v8bf;
typedef __attribute__((ext_vector_type(8)))  float  v8f;

union BF16x16 { v16bf v; v8bf h[2]; };

// Build a 16x32 bf16 A-fragment for one lane.
// Lanes 0-15 hold K {k0..k0+7, k0+16..k0+23}; lanes 16-31 hold {k0+8..15, k0+24..31}.
// 'p' must already point at row_base + k0 + half*8.
__device__ __forceinline__ v16bf load_afrag(const __bf16* p) {
    BF16x16 u;
    u.h[0] = *(const v8bf*)(p);
    u.h[1] = *(const v8bf*)(p + 16);
    return u.v;
}

// Async copy of 16 bytes global -> LDS (ASYNCcnt path, no VGPR data staging).
// The LDS operand is derived from the real shared-memory pointer so that
// (a) addr[31:0] is the raw LDS byte offset the instruction's VDST expects, and
// (b) the pointer ESCAPES into the asm, so the "memory" clobber forces the
//     compiler to treat the LDS tile as written (otherwise loads fold to undef).
__device__ __forceinline__ void async_g2l_b128(void* lds_ptr, const void* gptr) {
    unsigned ldsoff = (unsigned)(unsigned long long)lds_ptr;  // addr[31:0] = LDS offset
    unsigned long long ga = (unsigned long long)gptr;
    asm volatile("global_load_async_to_lds_b128 %0, %1, off"
                 :: "v"(ldsoff), "v"(ga) : "memory");
}
__device__ __forceinline__ void wait_asynccnt0() {
    asm volatile("s_wait_asynccnt 0x0" ::: "memory");
}

// ---------------- K0: weights fp32 -> bf16 ----------------
__global__ void k_weights(const float* tw, const float* pw, const float* gw,
                          const float* Ww, __bf16* wtpg, __bf16* wW) {
    int i = blockIdx.x * blockDim.x + threadIdx.x;
    if (i < 384 * 256) {
        int o = i >> 8, c = i & 255;
        float v = (o < 128) ? tw[o * 256 + c]
                : (o < 256) ? pw[(o - 128) * 256 + c]
                            : gw[(o - 256) * 256 + c];
        wtpg[i] = (__bf16)v;
    }
    if (i < 256 * 128) wW[i] = (__bf16)Ww[i];
}

// ---------------- K1: a = relu(bn1(x)) transposed to [B,N,C] bf16 ----------------
__global__ void k_bnrelu_t(const float* __restrict__ x,
                           const float* g1, const float* b1,
                           const float* m1, const float* v1,
                           __bf16* __restrict__ a_t) {
    __shared__ float tile[32][33];
    int b = blockIdx.z, c0 = blockIdx.y * 32, n0 = blockIdx.x * 32;
    int tx = threadIdx.x, ty = threadIdx.y; // 32 x 8
    #pragma unroll
    for (int i = 0; i < 4; ++i) {
        int c = c0 + ty + 8 * i;
        float inv = g1[c] * rsqrtf(v1[c] + 1e-5f);
        float sh  = b1[c] - m1[c] * inv;
        float v = x[((size_t)b * C_ + c) * N_ + n0 + tx];
        v = v * inv + sh;
        tile[ty + 8 * i][tx] = v > 0.f ? v : 0.f;
    }
    __syncthreads();
    #pragma unroll
    for (int i = 0; i < 4; ++i) {
        int n = n0 + ty + 8 * i;
        a_t[((size_t)b * N_ + n) * C_ + c0 + tx] = (__bf16)tile[tx][ty + 8 * i];
    }
}

// ---------------- K2: stacked conv GEMM: out[o,n] = wtpg[o,:] . a[:,n] + bias ----------------
// The 16(n)x256(K) activation tile is shared by all 8 waves of the block:
// stage it once in LDS with async global->LDS loads, then feed B-fragments via ds_load.
// o 0..127 -> theta, stored transposed [B,N,IC]; o 128..383 -> phi/g, stored [B,256,N]
__global__ void __launch_bounds__(256) k_conv(const __bf16* __restrict__ a_t,
                                              const __bf16* __restrict__ wtpg,
                                              const float* tb, const float* pb, const float* gb,
                                              __bf16* __restrict__ theta_t,
                                              __bf16* __restrict__ pg) {
    __shared__ __bf16 atile[16][C_];                 // 8 KB
    int b    = blockIdx.z;
    int wave = threadIdx.x >> 5, lane = threadIdx.x & 31;
    int lr   = lane & 15, half = lane >> 4;
    int o0   = (blockIdx.y * 8 + wave) * 16;
    int n0   = blockIdx.x * 16;

    // cooperative async stage: 256 threads x 32B = 8KB
    {
        int t    = threadIdx.x;
        int row  = t >> 4;            // 0..15
        int ch   = t & 15;            // 16B chunks, x2
        const __bf16* gp = a_t + ((size_t)b * N_ + n0 + row) * C_ + ch * 16;
        __bf16* lp = &atile[row][ch * 16];
        async_g2l_b128(lp,     gp);
        async_g2l_b128(lp + 8, gp + 8);
    }
    wait_asynccnt0();
    __syncthreads();

    const __bf16* wrow = wtpg + (size_t)(o0 + lr) * C_;
    const __bf16* brow = &atile[lr][0];

    v8f acc = {};
    #pragma unroll
    for (int k0 = 0; k0 < 256; k0 += 32) {
        v16bf af = load_afrag(wrow + k0 + half * 8);
        v16bf bf = load_afrag(brow + k0 + half * 8);   // ds_load_b128 x2
        acc = __builtin_amdgcn_wmma_f32_16x16x32_bf16(false, af, false, bf,
                                                      (short)0, acc, false, false);
    }

    const float* bias = (o0 < 128) ? (tb + o0)
                      : (o0 < 256) ? (pb + o0 - 128)
                                   : (gb + o0 - 256);
    int n = n0 + lr;
    if (o0 < 128) {
        v8bf outv;
        #pragma unroll
        for (int r = 0; r < 8; ++r) outv[r] = (__bf16)(acc[r] + bias[r + 8 * half]);
        *(v8bf*)(theta_t + ((size_t)b * N_ + n) * IC_ + o0 + 8 * half) = outv;
    } else {
        #pragma unroll
        for (int r = 0; r < 8; ++r) {
            int o = o0 + r + 8 * half;
            pg[((size_t)b * 256 + (o - 128)) * N_ + n] =
                (__bf16)(acc[r] + bias[r + 8 * half]);
        }
    }
}

// ---------------- K3: 2x2x2 maxpool; phi -> [B,MP,IC], g -> [B,IC,MP]; pad with 0 ----------------
__global__ void k_pool(const __bf16* __restrict__ pg,
                       __bf16* __restrict__ phi_t, __bf16* __restrict__ gmat) {
    int idx = blockIdx.x * blockDim.x + threadIdx.x;
    if (idx >= B_ * 256 * MP_) return;
    int m = idx % MP_;
    int j = (idx / MP_) % 256;
    int b = idx / (MP_ * 256);
    float v = 0.f;
    if (m < M_) {
        int mw = m % 14, mh = (m / 14) % 14, mt = m / 196;
        const __bf16* src = pg + ((size_t)b * 256 + j) * N_;
        float best = -3.4e38f;
        #pragma unroll
        for (int dt = 0; dt < 2; ++dt)
            #pragma unroll
            for (int dh = 0; dh < 2; ++dh)
                #pragma unroll
                for (int dw = 0; dw < 2; ++dw) {
                    int n = ((2 * mt + dt) * 28 + (2 * mh + dh)) * 28 + (2 * mw + dw);
                    float t = (float)src[n];
                    best = t > best ? t : best;
                }
        v = best;
    }
    if (j < 128) phi_t[((size_t)b * MP_ + m) * IC_ + j] = (__bf16)v;
    else         gmat [((size_t)b * IC_ + (j - 128)) * MP_ + m] = (__bf16)v;
}

// ---------------- K4: fused attention (f -> softmax -> y) + bn2+relu, wy -> [B,N,IC] bf16 ----------------
// Block: 128 threads (4 waves), 64 rows of N per block.
// Dynamic LDS: f (64x800 fp32 = 200KB) + p (64x800 bf16 = 100KB) = 300KB (<320KB/WGP)
__global__ void __launch_bounds__(128) k_attn(const __bf16* __restrict__ theta_t,
                                              const __bf16* __restrict__ phi_t,
                                              const __bf16* __restrict__ gmat,
                                              const float* g2, const float* b2,
                                              const float* m2, const float* v2,
                                              __bf16* __restrict__ wy_t) {
    extern __shared__ char smem_raw[];
    float*  f_lds = (float*)smem_raw;                        // [64][MP_]
    __bf16* p_lds = (__bf16*)(smem_raw + 64 * MP_ * 4);      // [64][MP_]

    int b    = blockIdx.y;
    int n0   = blockIdx.x * 64;
    int wave = threadIdx.x >> 5, lane = threadIdx.x & 31;
    int lr   = lane & 15, half = lane >> 4;

    // -------- phase 1: f = theta^T * phi --------
    const __bf16* trow = theta_t + ((size_t)b * N_ + n0 + wave * 16 + lr) * IC_;
    v16bf afr[4];
    #pragma unroll
    for (int k = 0; k < 4; ++k) afr[k] = load_afrag(trow + k * 32 + half * 8);

    for (int mt = 0; mt < 50; ++mt) {
        int m0 = mt * 16;
        const __bf16* prow = phi_t + ((size_t)b * MP_ + m0 + lr) * IC_;
        v8f acc = {};
        #pragma unroll
        for (int k = 0; k < 4; ++k) {
            v16bf bfr = load_afrag(prow + k * 32 + half * 8);
            acc = __builtin_amdgcn_wmma_f32_16x16x32_bf16(false, afr[k], false, bfr,
                                                          (short)0, acc, false, false);
        }
        #pragma unroll
        for (int r = 0; r < 8; ++r)
            f_lds[(wave * 16 + r + 8 * half) * MP_ + m0 + lr] = acc[r];
    }
    __syncthreads();

    // -------- phase 2: row softmax over the real M columns --------
    if (threadIdx.x < 64) {
        float* fr = f_lds + threadIdx.x * MP_;
        float mx = -3.4e38f;
        for (int m = 0; m < M_; ++m) mx = fr[m] > mx ? fr[m] : mx;
        float s = 0.f;
        for (int m = 0; m < M_; ++m) { float e = __expf(fr[m] - mx); fr[m] = e; s += e; }
        float inv = 1.f / s;
        __bf16* pr = p_lds + threadIdx.x * MP_;
        for (int m = 0; m < M_; ++m) pr[m] = (__bf16)(fr[m] * inv);
        for (int m = M_; m < MP_; ++m) pr[m] = (__bf16)0.f;   // padded cols contribute 0
    }
    __syncthreads();

    // -------- phase 3: y = att * g^T, fused bn2 + relu --------
    v8f accY[8];
    #pragma unroll
    for (int i = 0; i < 8; ++i) { v8f z = {}; accY[i] = z; }

    const __bf16* prow_l = p_lds + (wave * 16 + lr) * MP_;
    const __bf16* gbase  = gmat + ((size_t)b * IC_ + lr) * MP_ + half * 8;
    for (int m0 = 0; m0 < MP_; m0 += 32) {
        v16bf af = load_afrag(prow_l + m0 + half * 8);       // ds_load fragments
        if (m0 + 32 < MP_)
            __builtin_prefetch(gbase + m0 + 32, 0, 1);       // global_prefetch_b8 next K-step
        #pragma unroll
        for (int ict = 0; ict < 8; ++ict) {
            v16bf bfr = load_afrag(gbase + (size_t)(ict * 16) * MP_ + m0);
            accY[ict] = __builtin_amdgcn_wmma_f32_16x16x32_bf16(false, af, false, bfr,
                                                                (short)0, accY[ict],
                                                                false, false);
        }
    }

    #pragma unroll
    for (int ict = 0; ict < 8; ++ict) {
        int ic = ict * 16 + lr;
        float inv = g2[ic] * rsqrtf(v2[ic] + 1e-5f);
        float sh  = b2[ic] - m2[ic] * inv;
        #pragma unroll
        for (int r = 0; r < 8; ++r) {
            int n = n0 + wave * 16 + r + 8 * half;
            float val = accY[ict][r] * inv + sh;
            val = val > 0.f ? val : 0.f;
            wy_t[((size_t)b * N_ + n) * IC_ + ic] = (__bf16)val;
        }
    }
}

// ---------------- K5: z = W * wy + Wb + x (fp32 out) ----------------
// Same async-LDS staging trick for the shared 16(n)x128(K) wy tile.
__global__ void __launch_bounds__(256) k_final(const __bf16* __restrict__ wy_t,
                                               const __bf16* __restrict__ wW,
                                               const float* __restrict__ Wb,
                                               const float* __restrict__ x,
                                               float* __restrict__ out) {
    __shared__ __bf16 wtile[16][IC_];                // 4 KB
    int b    = blockIdx.z;
    int wave = threadIdx.x >> 5, lane = threadIdx.x & 31;
    int lr   = lane & 15, half = lane >> 4;
    int o0   = (blockIdx.y * 8 + wave) * 16;
    int n0   = blockIdx.x * 16;

    // cooperative async stage: 256 threads x 16B = 4KB
    {
        int t   = threadIdx.x;
        int row = t >> 4;             // 0..15
        int ch  = t & 15;             // 16B chunk
        const __bf16* gp = wy_t + ((size_t)b * N_ + n0 + row) * IC_ + ch * 8;
        async_g2l_b128(&wtile[row][ch * 8], gp);
    }
    wait_asynccnt0();
    __syncthreads();

    const __bf16* wrow = wW + (size_t)(o0 + lr) * IC_;
    const __bf16* brow = &wtile[lr][0];

    v8f acc = {};
    #pragma unroll
    for (int k0 = 0; k0 < 128; k0 += 32) {
        v16bf af = load_afrag(wrow + k0 + half * 8);
        v16bf bf = load_afrag(brow + k0 + half * 8);   // ds_load
        acc = __builtin_amdgcn_wmma_f32_16x16x32_bf16(false, af, false, bf,
                                                      (short)0, acc, false, false);
    }
    int n = n0 + lr;
    #pragma unroll
    for (int r = 0; r < 8; ++r) {
        int o = o0 + r + 8 * half;
        size_t idx = ((size_t)b * C_ + o) * N_ + n;
        out[idx] = acc[r] + Wb[o] + x[idx];
    }
}

// ---------------- host launcher ----------------
extern "C" void kernel_launch(void* const* d_in, const int* in_sizes, int n_in,
                              void* d_out, int out_size, void* d_ws, size_t ws_size,
                              hipStream_t stream) {
    const float* x   = (const float*)d_in[0];
    const float* g1  = (const float*)d_in[1];
    const float* b1  = (const float*)d_in[2];
    const float* m1  = (const float*)d_in[3];
    const float* v1  = (const float*)d_in[4];
    const float* tw  = (const float*)d_in[5];
    const float* tb  = (const float*)d_in[6];
    const float* pw  = (const float*)d_in[7];
    const float* pb  = (const float*)d_in[8];
    const float* gw  = (const float*)d_in[9];
    const float* gb  = (const float*)d_in[10];
    const float* g2  = (const float*)d_in[11];
    const float* b2  = (const float*)d_in[12];
    const float* m2  = (const float*)d_in[13];
    const float* v2  = (const float*)d_in[14];
    const float* Ww  = (const float*)d_in[15];
    const float* Wb  = (const float*)d_in[16];
    float* out = (float*)d_out;

    // workspace carve-up (bf16 intermediates, ~80 MB -> L2 resident)
    char* ws = (char*)d_ws;
    auto take = [&](size_t bytes) { char* p = ws; ws += (bytes + 255) & ~(size_t)255; return p; };
    __bf16* a_t     = (__bf16*)take((size_t)B_ * N_ * C_ * 2);
    __bf16* wtpg    = (__bf16*)take((size_t)384 * 256 * 2);
    __bf16* wWb16   = (__bf16*)take((size_t)256 * 128 * 2);
    __bf16* theta_t = (__bf16*)take((size_t)B_ * N_ * IC_ * 2);
    __bf16* pg      = (__bf16*)take((size_t)B_ * 256 * N_ * 2);
    __bf16* phi_t   = (__bf16*)take((size_t)B_ * MP_ * IC_ * 2);
    __bf16* gmat    = (__bf16*)take((size_t)B_ * IC_ * MP_ * 2);
    __bf16* wy_t    = (__bf16*)take((size_t)B_ * N_ * IC_ * 2);

    // K0: weight conversion
    k_weights<<<dim3((384 * 256 + 255) / 256), dim3(256), 0, stream>>>(
        tw, pw, gw, Ww, wtpg, wWb16);

    // K1: bn1 + relu + transpose to [B,N,C] bf16
    k_bnrelu_t<<<dim3(N_ / 32, C_ / 32, B_), dim3(32, 8), 0, stream>>>(
        x, g1, b1, m1, v1, a_t);

    // K2: stacked theta/phi/g GEMM (WMMA bf16, async-LDS staged B operand)
    k_conv<<<dim3(N_ / 16, 3, B_), dim3(256), 0, stream>>>(
        a_t, wtpg, tb, pb, gb, theta_t, pg);

    // K3: maxpool 2x2x2 with padding to MP_
    k_pool<<<dim3((B_ * 256 * MP_ + 255) / 256), dim3(256), 0, stream>>>(
        pg, phi_t, gmat);

    // K4: fused attention (WMMA -> softmax in LDS -> WMMA) + bn2 + relu
    size_t attn_lds = (size_t)64 * MP_ * 4 + (size_t)64 * MP_ * 2; // 307200 B
    k_attn<<<dim3(N_ / 64, B_), dim3(128), attn_lds, stream>>>(
        theta_t, phi_t, gmat, g2, b2, m2, v2, wy_t);

    // K5: final conv + bias + residual (fp32 out, async-LDS staged B operand)
    k_final<<<dim3(N_ / 16, 2, B_), dim3(256), 0, stream>>>(
        wy_t, wWb16, Wb, x, out);

    (void)in_sizes; (void)n_in; (void)out_size; (void)ws_size;
}
